// DecoderLayer_90967407329666
// MI455X (gfx1250) — matrix operand order repaired and checked by reading, not compiled, not run on hardware
//
#include <hip/hip_runtime.h>
#include <math.h>

// ---------------------------------------------------------------------------
// Problem constants
// ---------------------------------------------------------------------------
#define BB      2
#define TT      2048
#define DD      768
#define HH      8
#define HD      96
#define NAX     2
#define DRR     32
#define DFF     2048
#define NTYPE   16
#define BT      (BB*TT)          // 4096
#define D3      (3*DD)           // 2304
#define TE_W    (2*DD)           // 1536

typedef __attribute__((ext_vector_type(16))) __bf16 v16bf;
typedef __attribute__((ext_vector_type(8)))  float  v8f;

union BfVec { v16bf v; unsigned int u[8]; };

// Optional CDNA5 async global->LDS path (ASYNCcnt-tracked), guarded so the
// build cannot break if the toolchain lacks the builtins.
#if defined(__has_builtin)
# if __has_builtin(__builtin_amdgcn_global_load_async_to_lds_b128) && \
     __has_builtin(__builtin_amdgcn_s_wait_asynccnt)
#  define HAVE_ASYNC_LDS 1
# endif
#endif

#ifdef HAVE_ASYNC_LDS
// From the compiler diagnostic: param0 is 'int __vector_size__(16)' pointer in
// address space 1 (printed as "__device__"); param1 is the LDS address.
typedef int v4i_t __attribute__((vector_size(16)));
typedef __attribute__((address_space(1))) v4i_t as1_v4i;
typedef __attribute__((address_space(3))) v4i_t as3_v4i;
__device__ __forceinline__ void async_cp16B(const unsigned short* g, unsigned short* l) {
    // copy 16 bytes (8 bf16) global -> LDS, tracked by ASYNCcnt
    __builtin_amdgcn_global_load_async_to_lds_b128(
        (as1_v4i*)(unsigned short*)g, (as3_v4i*)l, 0, 0);
}
#endif

__device__ __forceinline__ unsigned short f2bf(float f) {
    unsigned int u = __float_as_uint(f);
    u += 0x7FFFu + ((u >> 16) & 1u);        // round-to-nearest-even
    return (unsigned short)(u >> 16);
}

__device__ __forceinline__ v8f zero8() {
    v8f z;
#pragma unroll
    for (int j = 0; j < 8; ++j) z[j] = 0.f;
    return z;
}

__device__ __forceinline__ v8f wmma_bf16(v16bf a, v16bf b, v8f c) {
    return __builtin_amdgcn_wmma_f32_16x16x32_bf16(false, a, false, b, (short)0, c, false, false);
}

// A-operand gather (16x32 bf16): lane L holds row (L&15); VGPR i holds packed
// K pair at k = (i<4 ? 2i : 2i+8) + 8*(L>=16).  (Vectorizes to 2x ds_load_b128.)
__device__ __forceinline__ v16bf ldA(const unsigned short* s, int row, int stride, int lh) {
    BfVec r;
#pragma unroll
    for (int i = 0; i < 8; ++i) {
        int k = ((i < 4) ? 2 * i : 2 * i + 8) + (lh << 3);
        r.u[i] = *(const unsigned int*)(s + row * stride + k);
    }
    return r.v;
}

// B-operand gather (32x16 bf16) from K-contiguous storage [n][k]:
// lane L holds column (L&15); VGPR i holds packed K pair at k = (i<4 ? 2i : 2i+8) + 8*(L>=16).
__device__ __forceinline__ v16bf ldB(const unsigned short* s, int row, int stride, int kbase, int lh) {
    BfVec r;
#pragma unroll
    for (int i = 0; i < 8; ++i) {
        int k = kbase + ((i < 4) ? 2 * i : 2 * i + 8) + (lh << 3);
        r.u[i] = *(const unsigned int*)(s + row * stride + k);
    }
    return r.v;
}

// ---------------------------------------------------------------------------
// fp32 [K][N] -> bf16 [N][K] transpose + convert (weights)
// ---------------------------------------------------------------------------
__global__ __launch_bounds__(256) void convT_kernel(const float* __restrict__ in,
                                                    unsigned short* __restrict__ out,
                                                    int K, int N) {
    __shared__ float t[32][33];
    int tx = threadIdx.x & 31, ty = threadIdx.x >> 5;
#pragma unroll
    for (int i = 0; i < 4; ++i) {
        int k = blockIdx.x * 32 + ty + i * 8;
        int n = blockIdx.y * 32 + tx;
        t[tx][ty + i * 8] = in[(size_t)k * N + n];
    }
    __syncthreads();
#pragma unroll
    for (int i = 0; i < 4; ++i) {
        int n = blockIdx.y * 32 + ty + i * 8;
        int k = blockIdx.x * 32 + tx;
        out[(size_t)n * K + k] = f2bf(t[ty + i * 8][tx]);
    }
}

// ---------------------------------------------------------------------------
// LayerNorm over D=768, output bf16
// ---------------------------------------------------------------------------
__global__ __launch_bounds__(256) void ln_kernel(const float* __restrict__ x,
                                                 const float* __restrict__ g,
                                                 const float* __restrict__ b,
                                                 unsigned short* __restrict__ out) {
    __shared__ float s1[256], s2[256];
    int row = blockIdx.x, tid = threadIdx.x;
    const float* xr = x + (size_t)row * DD;
    float v0 = xr[tid], v1 = xr[tid + 256], v2 = xr[tid + 512];
    s1[tid] = v0 + v1 + v2;
    s2[tid] = v0 * v0 + v1 * v1 + v2 * v2;
    __syncthreads();
    for (int o = 128; o > 0; o >>= 1) {
        if (tid < o) { s1[tid] += s1[tid + o]; s2[tid] += s2[tid + o]; }
        __syncthreads();
    }
    float mean = s1[0] * (1.f / DD);
    float var  = s2[0] * (1.f / DD) - mean * mean;
    float rstd = rsqrtf(var + 1e-5f);
    unsigned short* orow = out + (size_t)row * DD;
    orow[tid]       = f2bf((v0 - mean) * rstd * g[tid]       + b[tid]);
    orow[tid + 256] = f2bf((v1 - mean) * rstd * g[tid + 256] + b[tid + 256]);
    orow[tid + 512] = f2bf((v2 - mean) * rstd * g[tid + 512] + b[tid + 512]);
}

// ---------------------------------------------------------------------------
// GEMM stage loader: 128x32 tile of X[? x K] into sX, 16 bf16 per thread.
// ---------------------------------------------------------------------------
__device__ __forceinline__ void stage128(const unsigned short* __restrict__ X, int row0,
                                         int K, int k0, unsigned short* sX,
                                         int r, int c) {
#ifdef HAVE_ASYNC_LDS
    const unsigned short* gp = X + (size_t)(row0 + r) * K + k0 + c;
    async_cp16B(gp,     &sX[r * 32 + c]);
    async_cp16B(gp + 8, &sX[r * 32 + c + 8]);
#else
    const uint4* gp = (const uint4*)(X + (size_t)(row0 + r) * K + k0 + c);
    uint4 a0 = gp[0], a1 = gp[1];
    *(uint4*)&sX[r * 32 + c]     = a0;
    *(uint4*)&sX[r * 32 + c + 8] = a1;
#endif
}

__device__ __forceinline__ void stage_fence() {
#ifdef HAVE_ASYNC_LDS
    __builtin_amdgcn_s_wait_asynccnt(0);
#endif
    __syncthreads();
}

// ---------------------------------------------------------------------------
// GEMM: C[BT x N] = A[BT x K](bf16) @ Wt[N x K](bf16)
// 128x128 block, 8 waves (4M x 2N), wave tile 32x64 -> 8 WMMA / stage
// ---------------------------------------------------------------------------
__global__ __launch_bounds__(256) void gemm_qkv_kernel(const unsigned short* __restrict__ A,
                                                       const unsigned short* __restrict__ Wt,
                                                       float* __restrict__ C) {
    const int K = DD, N = D3;
    __shared__ unsigned short sA[128 * 32];
    __shared__ unsigned short sB[128 * 32];
    int tid = threadIdx.x, wid = tid >> 5, lane = tid & 31;
    int l15 = lane & 15, lh = lane >> 4;
    int m0 = blockIdx.x * 128, n0 = blockIdx.y * 128;
    int wm = (wid >> 1) * 32, wn = (wid & 1) * 64;
    int r = tid >> 1, c = (tid & 1) * 16;
    v8f acc[2][4];
#pragma unroll
    for (int i = 0; i < 2; ++i)
#pragma unroll
        for (int j = 0; j < 4; ++j) acc[i][j] = zero8();

    for (int k0 = 0; k0 < K; k0 += 32) {
        stage128(A,  m0, K, k0, sA, r, c);
        stage128(Wt, n0, K, k0, sB, r, c);
        if (k0 + 32 < K) {
            __builtin_prefetch(A  + (size_t)(m0 + r) * K + k0 + 32 + c, 0, 3);
            __builtin_prefetch(Wt + (size_t)(n0 + r) * K + k0 + 32 + c, 0, 3);
        }
        stage_fence();
        v16bf av[2], bv[4];
#pragma unroll
        for (int mi = 0; mi < 2; ++mi) av[mi] = ldA(sA, wm + mi * 16 + l15, 32, lh);
#pragma unroll
        for (int ni = 0; ni < 4; ++ni) bv[ni] = ldB(sB, wn + ni * 16 + l15, 32, 0, lh);
#pragma unroll
        for (int mi = 0; mi < 2; ++mi)
#pragma unroll
            for (int ni = 0; ni < 4; ++ni)
                acc[mi][ni] = wmma_bf16(av[mi], bv[ni], acc[mi][ni]);
        __syncthreads();
    }
#pragma unroll
    for (int mi = 0; mi < 2; ++mi)
#pragma unroll
        for (int ni = 0; ni < 4; ++ni)
#pragma unroll
            for (int j = 0; j < 8; ++j) {
                int row = m0 + wm + mi * 16 + j + (lh << 3);
                int col = n0 + wn + ni * 16 + l15;
                C[(size_t)row * N + col] = acc[mi][ni][j];
            }
}

// FC1: dual tile (a + gate) with fused SiLU epilogue -> bf16 [BT x DFF]
__global__ __launch_bounds__(256) void gemm_fc1_kernel(const unsigned short* __restrict__ A,
                                                       const unsigned short* __restrict__ Wt,
                                                       const float* __restrict__ bias,
                                                       unsigned short* __restrict__ ffin) {
    const int K = DD;
    __shared__ unsigned short sA[128 * 32];
    __shared__ unsigned short sBa[128 * 32];
    __shared__ unsigned short sBg[128 * 32];
    int tid = threadIdx.x, wid = tid >> 5, lane = tid & 31;
    int l15 = lane & 15, lh = lane >> 4;
    int m0 = blockIdx.x * 128, n0 = blockIdx.y * 128;
    int wm = (wid >> 1) * 32, wn = (wid & 1) * 64;
    int r = tid >> 1, c = (tid & 1) * 16;
    v8f accA[2][4], accG[2][4];
#pragma unroll
    for (int i = 0; i < 2; ++i)
#pragma unroll
        for (int j = 0; j < 4; ++j) { accA[i][j] = zero8(); accG[i][j] = zero8(); }

    for (int k0 = 0; k0 < K; k0 += 32) {
        stage128(A,  m0,       K, k0, sA,  r, c);
        stage128(Wt, n0,       K, k0, sBa, r, c);
        stage128(Wt, DFF + n0, K, k0, sBg, r, c);
        stage_fence();
        v16bf av[2], ba[4], bg[4];
#pragma unroll
        for (int mi = 0; mi < 2; ++mi) av[mi] = ldA(sA, wm + mi * 16 + l15, 32, lh);
#pragma unroll
        for (int ni = 0; ni < 4; ++ni) {
            ba[ni] = ldB(sBa, wn + ni * 16 + l15, 32, 0, lh);
            bg[ni] = ldB(sBg, wn + ni * 16 + l15, 32, 0, lh);
        }
#pragma unroll
        for (int mi = 0; mi < 2; ++mi)
#pragma unroll
            for (int ni = 0; ni < 4; ++ni) {
                accA[mi][ni] = wmma_bf16(av[mi], ba[ni], accA[mi][ni]);
                accG[mi][ni] = wmma_bf16(av[mi], bg[ni], accG[mi][ni]);
            }
        __syncthreads();
    }
#pragma unroll
    for (int mi = 0; mi < 2; ++mi)
#pragma unroll
        for (int ni = 0; ni < 4; ++ni)
#pragma unroll
            for (int j = 0; j < 8; ++j) {
                int row = m0 + wm + mi * 16 + j + (lh << 3);
                int col = n0 + wn + ni * 16 + l15;
                float a = accA[mi][ni][j] + bias[col];
                float g = accG[mi][ni][j] + bias[DFF + col];
                float s = g / (1.f + expf(-g));          // silu(gate)
                ffin[(size_t)row * DFF + col] = f2bf(s * a);
            }
}

// FC2: bf16 [BT x DFF] @ Wt[768 x 2048] + bias + residual -> d_out (f32)
__global__ __launch_bounds__(256) void gemm_fc2_kernel(const unsigned short* __restrict__ A,
                                                       const unsigned short* __restrict__ Wt,
                                                       const float* __restrict__ bias,
                                                       const float* __restrict__ xres,
                                                       float* __restrict__ out) {
    const int K = DFF, N = DD;
    __shared__ unsigned short sA[128 * 32];
    __shared__ unsigned short sB[128 * 32];
    int tid = threadIdx.x, wid = tid >> 5, lane = tid & 31;
    int l15 = lane & 15, lh = lane >> 4;
    int m0 = blockIdx.x * 128, n0 = blockIdx.y * 128;
    int wm = (wid >> 1) * 32, wn = (wid & 1) * 64;
    int r = tid >> 1, c = (tid & 1) * 16;
    v8f acc[2][4];
#pragma unroll
    for (int i = 0; i < 2; ++i)
#pragma unroll
        for (int j = 0; j < 4; ++j) acc[i][j] = zero8();

    for (int k0 = 0; k0 < K; k0 += 32) {
        stage128(A,  m0, K, k0, sA, r, c);
        stage128(Wt, n0, K, k0, sB, r, c);
        if (k0 + 32 < K) {
            __builtin_prefetch(A  + (size_t)(m0 + r) * K + k0 + 32 + c, 0, 3);
            __builtin_prefetch(Wt + (size_t)(n0 + r) * K + k0 + 32 + c, 0, 3);
        }
        stage_fence();
        v16bf av[2], bv[4];
#pragma unroll
        for (int mi = 0; mi < 2; ++mi) av[mi] = ldA(sA, wm + mi * 16 + l15, 32, lh);
#pragma unroll
        for (int ni = 0; ni < 4; ++ni) bv[ni] = ldB(sB, wn + ni * 16 + l15, 32, 0, lh);
#pragma unroll
        for (int mi = 0; mi < 2; ++mi)
#pragma unroll
            for (int ni = 0; ni < 4; ++ni)
                acc[mi][ni] = wmma_bf16(av[mi], bv[ni], acc[mi][ni]);
        __syncthreads();
    }
#pragma unroll
    for (int mi = 0; mi < 2; ++mi)
#pragma unroll
        for (int ni = 0; ni < 4; ++ni)
#pragma unroll
            for (int j = 0; j < 8; ++j) {
                int row = m0 + wm + mi * 16 + j + (lh << 3);
                int col = n0 + wn + ni * 16 + l15;
                size_t off = (size_t)row * N + col;
                out[off] = xres[off] + acc[mi][ni][j] + bias[col];
            }
}

// ---------------------------------------------------------------------------
// QKV post: type-emb add, RoPE, head layout [B,H,T,96], bf16 (q pre-scaled)
// ---------------------------------------------------------------------------
__global__ __launch_bounds__(256) void qkvpost_kernel(const float* __restrict__ qkv,
                                                      const int* __restrict__ x_type,
                                                      const int* __restrict__ seq_order,
                                                      const float* __restrict__ type_emb,
                                                      unsigned short* __restrict__ qb,
                                                      unsigned short* __restrict__ kb,
                                                      unsigned short* __restrict__ vb) {
    int idx = blockIdx.x * 256 + threadIdx.x;
    int j = idx % 48; int rest = idx / 48;
    int t = rest % TT; rest /= TT;
    int h = rest % HH; int b = rest / HH;

    size_t qkvrow = (size_t)(b * TT + t) * D3;
    int c = h * HD;
    int ty_q = x_type[b * (TT + 1) + t];
    int ty_k = x_type[b * (TT + 1) + t + 1];
    const float* teq = type_emb + (size_t)ty_q * TE_W;
    const float* tek = type_emb + (size_t)ty_k * TE_W + DD;
    const float invs = 0.10206207261596575f;   // 1/sqrt(96)
    size_t hrow = ((size_t)(b * HH + h) * TT + t) * HD;

    if (j < 32) {
        int axis = j >> 4, p = j & 15;
        int d0 = axis * DRR + 2 * p;
        float q0 = qkv[qkvrow + c + d0]     + teq[c + d0];
        float q1 = qkv[qkvrow + c + d0 + 1] + teq[c + d0 + 1];
        float k0 = qkv[qkvrow + DD + c + d0]     + tek[c + d0];
        float k1 = qkv[qkvrow + DD + c + d0 + 1] + tek[c + d0 + 1];
        int pq = seq_order[(size_t)(axis * BB + b) * (TT + 1) + t];
        int pk = seq_order[(size_t)(axis * BB + b) * (TT + 1) + t + 1];
        float invf = powf(10000.f, -(float)p / 16.f);
        float fq = (float)pq * invf, fk = (float)pk * invf;
        float cq = cosf(fq), sq = sinf(fq), ck = cosf(fk), sk = sinf(fk);
        qb[hrow + d0]     = f2bf((q0 * cq - q1 * sq) * invs);
        qb[hrow + d0 + 1] = f2bf((q1 * cq + q0 * sq) * invs);
        kb[hrow + d0]     = f2bf(k0 * ck - k1 * sk);
        kb[hrow + d0 + 1] = f2bf(k1 * ck + k0 * sk);
    } else {
        int d0 = NAX * DRR + 2 * (j - 32);
        qb[hrow + d0]     = f2bf((qkv[qkvrow + c + d0]     + teq[c + d0])     * invs);
        qb[hrow + d0 + 1] = f2bf((qkv[qkvrow + c + d0 + 1] + teq[c + d0 + 1]) * invs);
        kb[hrow + d0]     = f2bf(qkv[qkvrow + DD + c + d0]     + tek[c + d0]);
        kb[hrow + d0 + 1] = f2bf(qkv[qkvrow + DD + c + d0 + 1] + tek[c + d0 + 1]);
    }
    int dv = 2 * j;
    vb[hrow + dv]     = f2bf(qkv[qkvrow + 2 * DD + c + dv]);
    vb[hrow + dv + 1] = f2bf(qkv[qkvrow + 2 * DD + c + dv + 1]);
}

// ---------------------------------------------------------------------------
// Flash attention: 4 waves, 64 q-rows per block, k-tiles of 32, WMMA bf16
// Writes x = x_value + attn_out (f32)
// ---------------------------------------------------------------------------
__global__ __launch_bounds__(128) void attn_kernel(const unsigned short* __restrict__ qb,
                                                   const unsigned short* __restrict__ kb,
                                                   const unsigned short* __restrict__ vb,
                                                   const float* __restrict__ xval,
                                                   float* __restrict__ xout) {
    __shared__ unsigned short sK[32 * HD];     // [n][d]
    __shared__ unsigned short sV[HD * 32];     // [d][n] (transposed)
    __shared__ unsigned short sP[4][16 * 32];  // per-wave P tile
    int tid = threadIdx.x, wid = tid >> 5, lane = tid & 31;
    int l15 = lane & 15, lh = lane >> 4;
    int q0 = blockIdx.x * 64;
    int bh = blockIdx.y;
    int b = bh >> 3, h = bh & 7;

    // Q rows -> A registers (3 k-chunks of 32 over HD=96)
    v16bf aq[3];
    {
        int qrow = q0 + wid * 16 + l15;
        const unsigned short* qp = qb + ((size_t)bh * TT + qrow) * HD;
#pragma unroll
        for (int dd = 0; dd < 3; ++dd) {
            BfVec r;
#pragma unroll
            for (int i = 0; i < 8; ++i) {
                int k = ((i < 4) ? 2 * i : 2 * i + 8) + (lh << 3);
                r.u[i] = *(const unsigned int*)(qp + dd * 32 + k);
            }
            aq[dd] = r.v;
        }
    }

    v8f acc_o[6];
#pragma unroll
    for (int dt = 0; dt < 6; ++dt) acc_o[dt] = zero8();
    float mrow[8], lrow[8];
#pragma unroll
    for (int j = 0; j < 8; ++j) { mrow[j] = -1e30f; lrow[j] = 0.f; }

    int rbase = q0 + wid * 16 + (lh << 3);
    int nkt = (q0 + 64) >> 5;

    for (int kt = 0; kt < nkt; ++kt) {
        __syncthreads();
        {   // cooperative K/V tile load (128 threads, 32 rows x 96)
            int n = tid >> 2, ch = tid & 3;
            const uint4* ks = (const uint4*)(kb + ((size_t)bh * TT + kt * 32 + n) * HD + ch * 24);
            uint4 k0v = ks[0], k1v = ks[1], k2v = ks[2];
            *(uint4*)&sK[n * HD + ch * 24]      = k0v;
            *(uint4*)&sK[n * HD + ch * 24 + 8]  = k1v;
            *(uint4*)&sK[n * HD + ch * 24 + 16] = k2v;
            const uint4* vs = (const uint4*)(vb + ((size_t)bh * TT + kt * 32 + n) * HD + ch * 24);
            uint4 v0v = vs[0], v1v = vs[1], v2v = vs[2];
            __align__(16) unsigned short tmp[24];
            *(uint4*)(tmp)      = v0v;
            *(uint4*)(tmp + 8)  = v1v;
            *(uint4*)(tmp + 16) = v2v;
#pragma unroll
            for (int e = 0; e < 24; ++e) sV[(ch * 24 + e) * 32 + n] = tmp[e];
        }
        __syncthreads();

        if (kt * 32 > q0 + wid * 16 + 15) continue;   // fully masked for this wave

        // S = Q @ K^T  (two 16x16 tiles over the 32 keys)
        v8f s0 = zero8(), s1 = zero8();
#pragma unroll
        for (int dd = 0; dd < 3; ++dd) {
            v16bf b0 = ldB(sK, l15,      HD, dd * 32, lh);
            v16bf b1 = ldB(sK, 16 + l15, HD, dd * 32, lh);
            s0 = wmma_bf16(aq[dd], b0, s0);
            s1 = wmma_bf16(aq[dd], b1, s1);
        }

        int col0 = kt * 32 + l15;
        int col1 = col0 + 16;
        float alpha[8];
#pragma unroll
        for (int j = 0; j < 8; ++j) {
            int rr = rbase + j;
            float a = (col0 <= rr) ? s0[j] : -1e30f;
            float cc = (col1 <= rr) ? s1[j] : -1e30f;
            float mx = fmaxf(a, cc);
#pragma unroll
            for (int m = 1; m < 16; m <<= 1) mx = fmaxf(mx, __shfl_xor(mx, m, 32));
            float mn = fmaxf(mrow[j], mx);
            alpha[j] = expf(mrow[j] - mn);
            mrow[j] = mn;
            float p0 = expf(a - mn);
            float p1 = expf(cc - mn);
            float rs = p0 + p1;
#pragma unroll
            for (int m = 1; m < 16; m <<= 1) rs += __shfl_xor(rs, m, 32);
            lrow[j] = lrow[j] * alpha[j] + rs;
            int rl = j + (lh << 3);
            sP[wid][rl * 32 + l15]      = f2bf(p0);
            sP[wid][rl * 32 + 16 + l15] = f2bf(p1);
        }
#pragma unroll
        for (int dt = 0; dt < 6; ++dt)
#pragma unroll
            for (int j = 0; j < 8; ++j) acc_o[dt][j] *= alpha[j];

        // O += P @ V
        v16bf ap = ldA(&sP[wid][0], l15, 32, lh);
#pragma unroll
        for (int dt = 0; dt < 6; ++dt) {
            v16bf bv = ldB(sV, dt * 16 + l15, 32, 0, lh);
            acc_o[dt] = wmma_bf16(ap, bv, acc_o[dt]);
        }
    }

    // x = x_value + O/l
#pragma unroll
    for (int dt = 0; dt < 6; ++dt)
#pragma unroll
        for (int j = 0; j < 8; ++j) {
            int rr = rbase + j;
            int cc = h * HD + dt * 16 + l15;
            size_t off = (size_t)(b * TT + rr) * DD + cc;
            xout[off] = xval[off] + acc_o[dt][j] / lrow[j];
        }
}

// ---------------------------------------------------------------------------
// Host launcher
// ---------------------------------------------------------------------------
extern "C" void kernel_launch(void* const* d_in, const int* in_sizes, int n_in,
                              void* d_out, int out_size, void* d_ws, size_t ws_size,
                              hipStream_t stream) {
    (void)in_sizes; (void)n_in; (void)out_size; (void)ws_size;
    const int*   x_type   = (const int*)  d_in[0];
    const float* x_value  = (const float*)d_in[1];
    const int*   seq_ord  = (const int*)  d_in[2];
    const float* W_attn   = (const float*)d_in[3];
    const float* type_emb = (const float*)d_in[4];
    const float* g1       = (const float*)d_in[5];
    const float* b1       = (const float*)d_in[6];
    const float* g2       = (const float*)d_in[7];
    const float* b2       = (const float*)d_in[8];
    const float* W_fc1    = (const float*)d_in[9];
    const float* b_fc1    = (const float*)d_in[10];
    const float* W_fc2    = (const float*)d_in[11];
    const float* b_fc2    = (const float*)d_in[12];
    float* out = (float*)d_out;

    char* ws = (char*)d_ws;
    size_t off = 0;
    auto alloc = [&](size_t bytes) -> void* {
        void* p = ws + off;
        off = (off + bytes + 255) & ~(size_t)255;
        return p;
    };
    unsigned short* wattnT = (unsigned short*)alloc((size_t)D3 * DD * 2);
    unsigned short* wfc1T  = (unsigned short*)alloc((size_t)2 * DFF * DD * 2);
    unsigned short* wfc2T  = (unsigned short*)alloc((size_t)DD * DFF * 2);
    unsigned short* xhat   = (unsigned short*)alloc((size_t)BT * DD * 2);
    float*          qkv    = (float*)        alloc((size_t)BT * D3 * 4);
    unsigned short* qbuf   = (unsigned short*)alloc((size_t)BB * HH * TT * HD * 2);
    unsigned short* kbuf   = (unsigned short*)alloc((size_t)BB * HH * TT * HD * 2);
    unsigned short* vbuf   = (unsigned short*)alloc((size_t)BB * HH * TT * HD * 2);
    float*          xres   = (float*)        alloc((size_t)BT * DD * 4);
    unsigned short* ffin   = (unsigned short*)alloc((size_t)BT * DFF * 2);

    // Weight convert/transpose to bf16 [N][K]
    convT_kernel<<<dim3(DD / 32, D3 / 32),       256, 0, stream>>>(W_attn, wattnT, DD, D3);
    convT_kernel<<<dim3(DD / 32, 2 * DFF / 32),  256, 0, stream>>>(W_fc1,  wfc1T,  DD, 2 * DFF);
    convT_kernel<<<dim3(DFF / 32, DD / 32),      256, 0, stream>>>(W_fc2,  wfc2T,  DFF, DD);

    // LN1 -> bf16
    ln_kernel<<<BT, 256, 0, stream>>>(x_value, g1, b1, xhat);

    // QKV GEMM
    gemm_qkv_kernel<<<dim3(BT / 128, D3 / 128), 256, 0, stream>>>(xhat, wattnT, qkv);

    // type-emb + RoPE + head layout
    qkvpost_kernel<<<(BB * HH * TT * 48) / 256, 256, 0, stream>>>(
        qkv, x_type, seq_ord, type_emb, qbuf, kbuf, vbuf);

    // flash attention + residual
    attn_kernel<<<dim3(TT / 64, BB * HH), 128, 0, stream>>>(qbuf, kbuf, vbuf, x_value, xres);

    // LN2 -> bf16 (reuse xhat)
    ln_kernel<<<BT, 256, 0, stream>>>(xres, g2, b2, xhat);

    // FC1 + SiLU gate
    gemm_fc1_kernel<<<dim3(BT / 128, DFF / 128), 256, 0, stream>>>(xhat, wfc1T, b_fc1, ffin);

    // FC2 + bias + residual -> out
    gemm_fc2_kernel<<<dim3(BT / 128, DD / 128), 256, 0, stream>>>(ffin, wfc2T, b_fc2, xres, out);
}
